// RegionAugmentedSelfAttention_37563783970982
// MI455X (gfx1250) — matrix operand order repaired and checked by reading, not compiled, not run on hardware
//
#include <hip/hip_runtime.h>
#include <hip/hip_bf16.h>

// ---------------------------------------------------------------------------
// RegionAugmentedSelfAttention for MI455X (gfx1250, wave32, WMMA bf16)
// B=128, S=144 (12x12 grid), D=1024
// ---------------------------------------------------------------------------

typedef __attribute__((ext_vector_type(16))) __bf16 v16bf;
typedef __attribute__((ext_vector_type(8)))  float  v8f;

namespace cfg {
constexpr int B = 128;
constexpr int S = 144;
constexpr int D = 1024;
constexpr int GRID = 12;
constexpr int M = B * S;          // 18432 rows for the fused projections
constexpr int SPAD = 160;         // probs K-dim padded to multiple of 32
}

// ---------------------------------------------------------------------------
// Async global->LDS copy (GLOBAL_LOAD_ASYNC_TO_LDS_B128, ASYNCcnt-tracked).
// Builtin signature (from compiler diagnostic):
//   (int4 addrspace(1)* gsrc, int4 addrspace(3)* ldst, imm offset, imm cpol)
// Falls back to a synchronous uint4 copy if the builtin is unavailable.
// ---------------------------------------------------------------------------
#if defined(__has_builtin)
#if __has_builtin(__builtin_amdgcn_global_load_async_to_lds_b128)
#define HAVE_ASYNC_LDS 1
#endif
#endif

typedef int b128_t __attribute__((vector_size(16)));
typedef __attribute__((address_space(1))) b128_t gas_b128;
typedef __attribute__((address_space(3))) b128_t las_b128;

__device__ __forceinline__ void async_copy16(const void* g, void* l) {
#ifdef HAVE_ASYNC_LDS
  __builtin_amdgcn_global_load_async_to_lds_b128(
      (gas_b128*)g, (las_b128*)l, 0, 0);
#else
  *(uint4*)l = *(const uint4*)g;
#endif
}

__device__ __forceinline__ void async_wait_all() {
#ifdef HAVE_ASYNC_LDS
#if __has_builtin(__builtin_amdgcn_s_wait_asynccnt)
  __builtin_amdgcn_s_wait_asynccnt(0);
#else
  asm volatile("s_wait_asynccnt 0x0" ::: "memory");
#endif
#endif
}

// ---------------------------------------------------------------------------
// Kernel 1: positional bias  bias[i][j] = relu( relu(pos@Wl+bl) . w_g[i] )
// pos = [dx, dy, 0, 0] analytically (unit grid: w == h == 2 -> dw = dh = 0)
// One wave per (i,j) pair; 1024-wide dot via lanes + shuffle reduction.
// ---------------------------------------------------------------------------
__global__ __launch_bounds__(128)
void bias_kernel(const float* __restrict__ Wl, const float* __restrict__ bl,
                 const float* __restrict__ wg, float* __restrict__ biasB) {
  const int pair = blockIdx.x * 4 + (threadIdx.x >> 5);
  const int lane = threadIdx.x & 31;
  if (pair >= cfg::S * cfg::S) return;
  const int i = pair / cfg::S;
  const int j = pair % cfg::S;

  const float cxi = (float)(i / cfg::GRID) + 0.5f;
  const float cyi = (float)(i % cfg::GRID) + 0.5f;
  const float cxj = (float)(j / cfg::GRID) + 0.5f;
  const float cyj = (float)(j % cfg::GRID) + 0.5f;
  const float dx = logf(fmaxf(fabsf((cxi - cxj) * 0.5f), 1e-10f));
  const float dy = logf(fmaxf(fabsf((cyi - cyj) * 0.5f), 1e-10f));
  // dw = log(2/2) = 0, dh = 0 -> Wl rows 2,3 contribute nothing

  float acc = 0.0f;
  for (int d = lane; d < cfg::D; d += 32) {
    float rel = dx * Wl[d] + dy * Wl[cfg::D + d] + bl[d];
    rel = fmaxf(rel, 0.0f);
    acc += rel * wg[(size_t)i * cfg::D + d];
  }
  #pragma unroll
  for (int off = 16; off > 0; off >>= 1) acc += __shfl_xor(acc, off);
  if (lane == 0) biasB[pair] = fmaxf(acc, 0.0f);
}

// ---------------------------------------------------------------------------
// Kernel 2: y = relu(X @ W + b) as bf16.  X:(18432,1024) f32, W:(1024,1024) f32.
// Tile 128x64x32, 8 waves (4x2), each wave 32x32 via 2x2 WMMA bf16 tiles.
// fp32 -> bf16 conversion happens while staging tiles into LDS (so the
// synchronous VGPR path is required here).
// ---------------------------------------------------------------------------
__global__ __launch_bounds__(256)
void qkv_proj_kernel(const float* __restrict__ X, const float* __restrict__ W,
                     const float* __restrict__ bias, __bf16* __restrict__ Y) {
  constexpr int BM = 128, BN = 64, BK = 32;
  constexpr int LDA = 40, LDB = 40;          // row pad -> 80B stride, 16B aligned
  __shared__ __bf16 la[BM][LDA];             // la[m][k]
  __shared__ __bf16 lb[BN][LDB];             // lb[n][k]  (W tile transposed)

  const int tid   = threadIdx.x;
  const int lane  = tid & 31;
  const int wid   = tid >> 5;
  const int waveM = wid >> 1;                // 0..3
  const int waveN = wid & 1;                 // 0..1

  const int nt = blockIdx.x & 15;            // 1024/64 = 16 N tiles
  const int mt = blockIdx.x >> 4;            // 18432/128 = 144 M tiles
  const int m0 = mt * BM;
  const int n0 = nt * BN;

  v8f acc[2][2] = {};
  const int kh  = (lane < 16) ? 0 : 8;       // ISA 16-bit A layout
  const int kh2 = (lane < 16) ? 0 : 16;      // unused: B is read with kh, same k order as A

  for (int k0 = 0; k0 < cfg::D; k0 += BK) {
    // A tile: 128x32 f32, 1024 float4 slots over 256 threads (4 rounds)
    #pragma unroll
    for (int s = 0; s < 4; ++s) {
      const int idx = s * 256 + tid;
      const int r = idx >> 3, seg = idx & 7;
      const float4 f = *(const float4*)(X + (size_t)(m0 + r) * cfg::D + k0 + seg * 4);
      __bf16* dst = &la[r][seg * 4];
      dst[0] = (__bf16)f.x; dst[1] = (__bf16)f.y;
      dst[2] = (__bf16)f.z; dst[3] = (__bf16)f.w;
    }
    // B tile: W[k0..k0+31][n0..n0+63] -> lb[n][k] (transposed)
    #pragma unroll
    for (int s = 0; s < 2; ++s) {
      const int idx = s * 256 + tid;
      const int r = idx >> 4, seg = idx & 15;      // r = k, seg*4 = n
      const float4 f = *(const float4*)(W + (size_t)(k0 + r) * cfg::D + n0 + seg * 4);
      lb[seg * 4 + 0][r] = (__bf16)f.x;
      lb[seg * 4 + 1][r] = (__bf16)f.y;
      lb[seg * 4 + 2][r] = (__bf16)f.z;
      lb[seg * 4 + 3][r] = (__bf16)f.w;
    }
    __syncthreads();

    #pragma unroll
    for (int tm = 0; tm < 2; ++tm) {
      v16bf a;
      const int rr = waveM * 32 + tm * 16 + (lane & 15);
      #pragma unroll
      for (int e = 0; e < 8; ++e) {
        a[e]     = la[rr][kh + e];
        a[e + 8] = la[rr][kh + 16 + e];
      }
      #pragma unroll
      for (int tn = 0; tn < 2; ++tn) {
        v16bf bb;
        const int cc = waveN * 32 + tn * 16 + (lane & 15);
        #pragma unroll
        for (int e = 0; e < 8; ++e) { bb[e] = lb[cc][kh + e]; bb[e + 8] = lb[cc][kh + 16 + e]; }
        acc[tm][tn] = __builtin_amdgcn_wmma_f32_16x16x32_bf16(
            false, a, false, bb, (short)0, acc[tm][tn], false, false);
      }
    }
    __syncthreads();
  }

  // epilogue: bias + relu, store bf16
  #pragma unroll
  for (int tn = 0; tn < 2; ++tn) {
    const int col = n0 + waveN * 32 + tn * 16 + (lane & 15);
    const float bv = bias[col];
    #pragma unroll
    for (int tm = 0; tm < 2; ++tm) {
      #pragma unroll
      for (int r = 0; r < 8; ++r) {
        const int roff = (lane < 16) ? r : (r + 8);
        const int row  = m0 + waveM * 32 + tm * 16 + roff;
        const float v  = fmaxf(acc[tm][tn][r] + bv, 0.0f);
        Y[(size_t)row * cfg::D + col] = (__bf16)v;
      }
    }
  }
}

// ---------------------------------------------------------------------------
// Kernel 3: fused attention scores + softmax.
// Block = (batch, 16-row strip). 9 waves, wave w owns the 16x16 j-tile w.
// q/k tiles are bf16 already -> staged into LDS with async global->LDS B128.
// scores = scale * q k^T + bias -> softmax over all 144 j -> bf16 probs,
// written with row stride SPAD=160 and zero-padded K tail.
// ---------------------------------------------------------------------------
__global__ __launch_bounds__(288)
void attn_softmax_kernel(const __bf16* __restrict__ qb, const __bf16* __restrict__ kb,
                         const float* __restrict__ biasB, __bf16* __restrict__ probs) {
  constexpr int LDQ = 40, LDK = 40, LSC = 152;
  __shared__ __bf16 lq[16][LDQ];
  __shared__ __bf16 lk[cfg::S][LDK];
  __shared__ float  scores[16][LSC];
  __shared__ float  pmax[16][18];
  __shared__ float  psum[16][18];
  __shared__ float  rowmax[16];
  __shared__ float  rowsum[16];

  const int tid  = threadIdx.x;
  const int lane = tid & 31;
  const int w    = tid >> 5;                 // 0..8 -> j tile

  const int rt = blockIdx.x % 9;
  const int b  = blockIdx.x / 9;
  const int i0 = rt * 16;
  const size_t qRow0 = (size_t)b * cfg::S + i0;
  const size_t kRow0 = (size_t)b * cfg::S;

  v8f acc = {};
  const int kh  = (lane < 16) ? 0 : 8;
  const int kh2 = (lane < 16) ? 0 : 16;
  const int jg  = w * 16 + (lane & 15);      // global column this lane owns

  for (int k0 = 0; k0 < cfg::D; k0 += 32) {
    if (tid < 64) {                          // q tile: 16 rows x 4 x 16B
      const int r = tid >> 2, seg = tid & 3;
      async_copy16(qb + (qRow0 + r) * cfg::D + k0 + seg * 8, &lq[r][seg * 8]);
    }
    #pragma unroll
    for (int s = 0; s < 2; ++s) {            // k tile: 144 rows x 4 x 16B
      const int idx = s * 288 + tid;
      const int r = idx >> 2, seg = idx & 3;
      async_copy16(kb + (kRow0 + r) * cfg::D + k0 + seg * 8, &lk[r][seg * 8]);
    }
    async_wait_all();
    __syncthreads();

    v16bf a, bb;
    const int ar = lane & 15;
    #pragma unroll
    for (int e = 0; e < 8; ++e) {
      a[e]     = lq[ar][kh + e];
      a[e + 8] = lq[ar][kh + 16 + e];
    }
    #pragma unroll
    for (int e = 0; e < 8; ++e) { bb[e] = lk[jg][kh + e]; bb[e + 8] = lk[jg][kh + 16 + e]; }
    acc = __builtin_amdgcn_wmma_f32_16x16x32_bf16(
        false, a, false, bb, (short)0, acc, false, false);
    __syncthreads();
  }

  const float scale = 0.03125f;              // 1024^-0.5
  #pragma unroll
  for (int r = 0; r < 8; ++r) {
    const int roff = (lane < 16) ? r : (r + 8);
    scores[roff][jg] = acc[r] * scale + biasB[(i0 + roff) * cfg::S + jg];
  }
  __syncthreads();

  // softmax: 288 threads = 16 rows x 18 chunks of 8 columns
  const int srow  = tid / 18;
  const int chunk = tid % 18;
  {
    float m = -3.4e38f;
    #pragma unroll
    for (int c = 0; c < 8; ++c) m = fmaxf(m, scores[srow][chunk * 8 + c]);
    pmax[srow][chunk] = m;
  }
  __syncthreads();
  if (tid < 16) {
    float m = pmax[tid][0];
    for (int c = 1; c < 18; ++c) m = fmaxf(m, pmax[tid][c]);
    rowmax[tid] = m;
  }
  __syncthreads();
  {
    const float m = rowmax[srow];
    float s = 0.0f;
    #pragma unroll
    for (int c = 0; c < 8; ++c) {
      const float e = __expf(scores[srow][chunk * 8 + c] - m);
      scores[srow][chunk * 8 + c] = e;
      s += e;
    }
    psum[srow][chunk] = s;
  }
  __syncthreads();
  if (tid < 16) {
    float s = psum[tid][0];
    for (int c = 1; c < 18; ++c) s += psum[tid][c];
    rowsum[tid] = s;
  }
  __syncthreads();
  {
    const float inv = 1.0f / rowsum[srow];
    const size_t pRow = ((size_t)b * cfg::S + i0 + srow) * cfg::SPAD;
    #pragma unroll
    for (int c = 0; c < 8; ++c) {
      const int col = chunk * 8 + c;
      probs[pRow + col] = (__bf16)(scores[srow][col] * inv);
    }
  }
  if (tid < 256) {                           // zero K padding cols 144..159
    const int r = tid >> 4, pc = cfg::S + (tid & 15);
    probs[((size_t)b * cfg::S + i0 + r) * cfg::SPAD + pc] = (__bf16)0.0f;
  }
}

// ---------------------------------------------------------------------------
// Kernel 4: out = probs @ v.  Per batch: (144 x 160pad) @ (144 x 1024) -> f32.
// Tile 48x128x32, 8 waves each owning a 48x16 column strip (3 WMMA row tiles).
// A (probs) tile is a straight bf16 copy -> async global->LDS; the v tile
// needs a transpose so it stays on the VGPR + ds_store path.
// ---------------------------------------------------------------------------
__global__ __launch_bounds__(256)
void out_gemm_kernel(const __bf16* __restrict__ probs, const __bf16* __restrict__ vb,
                     float* __restrict__ out) {
  constexpr int BM = 48, BN = 128, BK = 32;
  constexpr int LDA = 40, LDB = 40;
  __shared__ __bf16 la[BM][LDA];             // probs tile [i][j]
  __shared__ __bf16 lv[BN][LDB];             // v tile transposed: lv[d][j]

  const int tid  = threadIdx.x;
  const int lane = tid & 31;
  const int w    = tid >> 5;                 // 0..7 -> d tile

  int bx = blockIdx.x;
  const int nt = bx & 7;  bx >>= 3;          // 1024/128 = 8
  const int mt = bx % 3;                     // 144/48  = 3
  const int b  = bx / 3;
  const int i0 = mt * BM;
  const int d0 = nt * BN;

  v8f acc[3] = {};
  const int kh  = (lane < 16) ? 0 : 8;
  const int kh2 = (lane < 16) ? 0 : 16;
  const int cl  = w * 16 + (lane & 15);      // local col in lv

  for (int j0 = 0; j0 < cfg::SPAD; j0 += BK) {
    if (tid < 192) {                         // A: 48 rows x 4 x 16B (async)
      const int r = tid >> 2, seg = tid & 3;
      async_copy16(probs + ((size_t)b * cfg::S + i0 + r) * cfg::SPAD + j0 + seg * 8,
                   &la[r][seg * 8]);
    }
    #pragma unroll
    for (int s = 0; s < 2; ++s) {            // B: v rows j0..j0+31 x 128 cols
      const int idx = s * 256 + tid;
      const int r = idx >> 4, seg = idx & 15;        // r = j, seg*8 = d
      uint4 u;
      if (j0 + r < cfg::S) {
        u = *(const uint4*)(vb + ((size_t)b * cfg::S + j0 + r) * cfg::D + d0 + seg * 8);
      } else {
        u.x = 0u; u.y = 0u; u.z = 0u; u.w = 0u;      // padded K rows -> zeros
      }
      const __bf16* tp = (const __bf16*)&u;
      #pragma unroll
      for (int q = 0; q < 8; ++q) lv[seg * 8 + q][r] = tp[q];
    }
    async_wait_all();
    __syncthreads();

    v16bf bb;
    #pragma unroll
    for (int e = 0; e < 8; ++e) { bb[e] = lv[cl][kh + e]; bb[e + 8] = lv[cl][kh + 16 + e]; }
    #pragma unroll
    for (int tm = 0; tm < 3; ++tm) {
      v16bf a;
      const int rr = tm * 16 + (lane & 15);
      #pragma unroll
      for (int e = 0; e < 8; ++e) {
        a[e]     = la[rr][kh + e];
        a[e + 8] = la[rr][kh + 16 + e];
      }
      acc[tm] = __builtin_amdgcn_wmma_f32_16x16x32_bf16(
          false, a, false, bb, (short)0, acc[tm], false, false);
    }
    __syncthreads();
  }

  const int col = d0 + w * 16 + (lane & 15);
  #pragma unroll
  for (int tm = 0; tm < 3; ++tm) {
    #pragma unroll
    for (int r = 0; r < 8; ++r) {
      const int roff = (lane < 16) ? r : (r + 8);
      const int row  = i0 + tm * 16 + roff;
      out[((size_t)b * cfg::S + row) * cfg::D + col] = acc[tm][r];
    }
  }
}

// ---------------------------------------------------------------------------
// Host launch
// ---------------------------------------------------------------------------
extern "C" void kernel_launch(void* const* d_in, const int* in_sizes, int n_in,
                              void* d_out, int out_size, void* d_ws, size_t ws_size,
                              hipStream_t stream) {
  (void)in_sizes; (void)n_in; (void)out_size; (void)ws_size;

  const float* query = (const float*)d_in[0];
  const float* key   = (const float*)d_in[1];
  const float* value = (const float*)d_in[2];
  const float* Wq    = (const float*)d_in[3];
  const float* bq    = (const float*)d_in[4];
  const float* Wk    = (const float*)d_in[5];
  const float* bk    = (const float*)d_in[6];
  const float* Wv    = (const float*)d_in[7];
  const float* bv    = (const float*)d_in[8];
  const float* Wl    = (const float*)d_in[9];
  const float* bl    = (const float*)d_in[10];
  const float* wg    = (const float*)d_in[11];
  float* out = (float*)d_out;

  // workspace layout (bytes)
  const size_t QKV_BYTES   = (size_t)cfg::M * cfg::D * sizeof(__bf16);       // 36 MiB each
  const size_t PROBS_BYTES = (size_t)cfg::B * cfg::S * cfg::SPAD * sizeof(__bf16);
  char* ws = (char*)d_ws;
  __bf16* qb    = (__bf16*)(ws);
  __bf16* kb    = (__bf16*)(ws + QKV_BYTES);
  __bf16* vbuf  = (__bf16*)(ws + 2 * QKV_BYTES);
  __bf16* probs = (__bf16*)(ws + 3 * QKV_BYTES);
  float*  biasB = (float*)(ws + 3 * QKV_BYTES + PROBS_BYTES);

  // 1) positional bias (independent)
  bias_kernel<<<dim3((cfg::S * cfg::S) / 4), dim3(128), 0, stream>>>(Wl, bl, wg, biasB);

  // 2) q/k/v projections: 144 M-tiles x 16 N-tiles
  const int qkvBlocks = (cfg::M / 128) * (cfg::D / 64);
  qkv_proj_kernel<<<dim3(qkvBlocks), dim3(256), 0, stream>>>(query, Wq, bq, qb);
  qkv_proj_kernel<<<dim3(qkvBlocks), dim3(256), 0, stream>>>(key,   Wk, bk, kb);
  qkv_proj_kernel<<<dim3(qkvBlocks), dim3(256), 0, stream>>>(value, Wv, bv, vbuf);

  // 3) scores + softmax: one block per (batch, 16-row strip)
  attn_softmax_kernel<<<dim3(cfg::B * (cfg::S / 16)), dim3(288), 0, stream>>>(
      qb, kb, biasB, probs);

  // 4) out = probs @ v : B * 3 * 8 blocks
  out_gemm_kernel<<<dim3(cfg::B * 3 * 8), dim3(256), 0, stream>>>(probs, vbuf, out);
}
